// BasicMambaLayer_77515569758968
// MI455X (gfx1250) — compile-verified, matrix operand order checked by reading, not run on hardware
//
#include <hip/hip_runtime.h>
#include <hip/hip_bf16.h>
#include <math.h>

namespace {

constexpr int DIM   = 96;
constexpr int DIc   = 192;   // DI
constexpr int DSc   = 16;    // DS
constexpr int DTR   = 6;
constexpr int DCc   = 4;     // conv taps
constexpr int NL    = 2;
constexpr int INDIM = 6144;  // DIM * 4*4*4
constexpr int L     = 4096;  // 16^3 tokens
constexpr int VOL   = 64 * 64 * 64;
constexpr float EPS = 1e-5f;

typedef __bf16 bf16_t;
typedef __attribute__((ext_vector_type(16))) __bf16 v16bf;
typedef __attribute__((ext_vector_type(8)))  __bf16 v8bf;
typedef __attribute__((ext_vector_type(8)))  float  v8f;
typedef __attribute__((ext_vector_type(4)))  unsigned int v4u;
typedef __attribute__((ext_vector_type(8)))  int v8i;
typedef __attribute__((ext_vector_type(4)))  int v4i;

#if defined(__has_builtin)
#if __has_builtin(__builtin_amdgcn_tensor_load_to_lds)
#define HAVE_TDM 1
#else
#define HAVE_TDM 0
#endif
#else
#define HAVE_TDM 0
#endif

__device__ __forceinline__ float silu_f(float x){ return x / (1.f + __expf(-x)); }
__device__ __forceinline__ float softplus_f(float x){ return (x > 20.f) ? x : log1pf(__expf(x)); }
__device__ __forceinline__ float gelu_f(float x){ return 0.5f * x * (1.f + erff(x * 0.70710678118654752f)); }

// K offset inside a 16x32 bf16 A/B fragment for packed element e (0..15), lane-half g (0/1).
__device__ __forceinline__ int kfrag(int e, int g){ return ((e & 8) << 1) + (g << 3) + (e & 7); }

// Gather index into x (C=96, 64,64,64) for permutation PERM, token t, feature f.
template<int PERM>
__device__ __forceinline__ int gidx(int t, int f){
  int a0 = t >> 8, a1 = (t >> 4) & 15, a2 = t & 15;
  int c = f >> 6, r = f & 63;
  int q0 = r >> 4, q1 = (r >> 2) & 3, q2 = r & 3;
  int hi, wi, di, p1, p2, p3;
  if constexpr (PERM == 0){ hi=a0; wi=a1; di=a2; p1=q0; p2=q1; p3=q2; }
  else if constexpr (PERM == 1){ wi=a0; di=a1; hi=a2; p2=q0; p3=q1; p1=q2; }
  else                        { di=a0; hi=a1; wi=a2; p3=q0; p1=q1; p2=q2; }
  return ((c << 6) + hi * 4 + p1) * 4096 + ((wi * 4 + p2) << 6) + (di * 4 + p3);
}

// LDS index of feature kk (0..63) / token-row mr inside the per-chunk 3D tile (x-innermost).
template<int PERM>
__device__ __forceinline__ int lidx(int kk, int mr){
  int q0 = kk >> 4, q1 = (kk >> 2) & 3, q2 = kk & 3;
  if constexpr (PERM == 0) return q0 * 256 + q1 * 64 + mr * 4 + q2;       // x=D(64) y=Wq z=Hq
  else if constexpr (PERM == 1) return (mr * 4 + q2) * 16 + q0 * 4 + q1;  // x=Dq y=Wq z=H(64)
  else return q1 * 256 + (mr * 4 + q2) * 4 + q0;                          // x=Dq y=W(64) z=Hq
}

#if HAVE_TDM
// Issue a 3D TDM tile load: tile (t0,t1,t2), element=4B, strides 64 / 4096 elements.
__device__ __forceinline__ void tdm_load_3d(unsigned lds_off, unsigned long long gaddr,
                                            int t0, int t1, int t2){
  v4u g0;
  g0[0] = 1u;                                              // count=1 (valid descriptor)
  g0[1] = lds_off;                                         // LDS byte address
  g0[2] = (unsigned)(gaddr & 0xffffffffull);               // global_addr[31:0]
  g0[3] = (unsigned)((gaddr >> 32) & 0x1ffffffull) | (2u << 30);  // addr[56:32] | type=2
  const unsigned td0 = 1u << 20, td1 = 1u << 20;           // generous OOB limits
  const unsigned long long S0 = 64ull, S1 = 4096ull;       // dim0/dim1 strides (elements)
  v8i g1;
  g1[0] = (int)(2u << 16);                                 // data_size=2 (4 bytes)
  g1[1] = (int)((td0 & 0xffffu) << 16);                    // tensor_dim0[15:0]
  g1[2] = (int)(((td0 >> 16) & 0xffffu) | ((td1 & 0xffffu) << 16));
  g1[3] = (int)(((td1 >> 16) & 0xffffu) | ((unsigned)(t0 & 0xffff) << 16)); // tile_dim0
  g1[4] = (int)(((unsigned)t1 & 0xffffu) | ((unsigned)t2 << 16));           // tile_dim1/2
  g1[5] = (int)(S0 & 0xffffffffull);
  g1[6] = (int)(((S0 >> 32) & 0xffffull) | ((S1 & 0xffffull) << 16));
  g1[7] = (int)((S1 >> 16) & 0xffffffffull);
  v4i g2; g2[0] = (int)(1u << 20); g2[1] = 0; g2[2] = 0; g2[3] = 0;  // tensor_dim2 big
  v4i g3; g3[0] = 0; g3[1] = 0; g3[2] = 0; g3[3] = 0;
  v8i g4; g4[0] = 0; g4[1] = 0; g4[2] = 0; g4[3] = 0;
          g4[4] = 0; g4[5] = 0; g4[6] = 0; g4[7] = 0;
  __builtin_amdgcn_tensor_load_to_lds(g0, g1, g2, g3, g4, 0);
}
#endif

// ---------------- weight packing (bf16, WMMA fragment order) ----------------
// dr pack: [kb(192)][nt(6)][lane(32)][e(16)]  from dr_W (6144,96)
// de pack: [kb(3)][nt(384)][lane(32)][e(16)]  from de_W (96,6144)
__global__ void k_pack(const float* __restrict__ drW, const float* __restrict__ deW,
                       bf16_t* __restrict__ drp, bf16_t* __restrict__ dep){
  int idx = blockIdx.x * 256 + threadIdx.x;
  if (idx >= INDIM * DIM) return;
  int e = idx & 15, lane = (idx >> 4) & 31, rem = idx >> 9;
  int g = lane >> 4, mr = lane & 15;
  int K = kfrag(e, g);
  { int nt = rem % 6, kb = rem / 6;
    drp[idx] = (bf16_t)drW[(kb * 32 + K) * DIM + nt * 16 + mr]; }
  { int nt = rem % 384, kb = rem / 384;
    dep[idx] = (bf16_t)deW[(kb * 32 + K) * INDIM + nt * 16 + mr]; }
}

__global__ void k_zero(float* __restrict__ p, int n){
  int idx = blockIdx.x * 256 + threadIdx.x;
  if (idx < n) p[idx] = 0.f;
}

// ---------------- dr GEMM: xf = x_perm @ dr_W (M=4096,N=96,K=6144) ----------------
// One block = one 16-token tile; 6 waves cover the 6 N-tiles; A tile staged in LDS
// by the Tensor Data Mover (3D strided tile, double-buffered, s_wait_tensorcnt).
template<int PERM>
__global__ void __launch_bounds__(192)
k_dr_gemm(const float* __restrict__ x, const bf16_t* __restrict__ drp,
          float* __restrict__ xf){
  __shared__ float sA[2][1024];          // only LDS object -> offset 0; 4KB per buffer
  const int tid  = threadIdx.x;
  const int wave = tid >> 5;             // n-tile 0..5
  const int lane = tid & 31;
  const int g    = lane >> 4;
  const int mr   = lane & 15;
  const int tile = blockIdx.x;           // 256 token tiles
  const int a0 = tile >> 4, a1 = tile & 15;

  long long base; int nx, ny;            // tile geometry (strides always 64 / 4096)
  if constexpr (PERM == 0){ base = (long long)a0 * 4 * 4096 + (long long)a1 * 4 * 64; nx = 64; ny = 4; }
  else if constexpr (PERM == 1){ base = (long long)a0 * 4 * 64 + (long long)a1 * 4;   nx = 4;  ny = 4; }
  else                        { base = (long long)a1 * 4 * 4096 + (long long)a0 * 4;  nx = 4;  ny = 64; }
  const int nz = 1024 / (nx * ny);

  v8f acc = {};

#if HAVE_TDM
  if (tid < 32)
    tdm_load_3d(0u, (unsigned long long)(size_t)x + 4ull * (unsigned long long)(base), nx, ny, nz);
  for (int c = 0; c < DIM; ++c){
    const int buf = c & 1;
    if (tid < 32){
      if (c + 1 < DIM){
        tdm_load_3d((unsigned)((buf ^ 1) * 4096),
                    (unsigned long long)(size_t)x +
                        4ull * ((unsigned long long)(c + 1) * 262144ull + (unsigned long long)base),
                    nx, ny, nz);
        __builtin_amdgcn_s_wait_tensorcnt(1);   // chunk c landed, chunk c+1 in flight
      } else {
        __builtin_amdgcn_s_wait_tensorcnt(0);
      }
    }
    __syncthreads();
#pragma unroll
    for (int s = 0; s < 2; ++s){
      v16bf a, b;
#pragma unroll
      for (int e = 0; e < 16; ++e)
        a[e] = (bf16_t)sA[buf][lidx<PERM>(s * 32 + kfrag(e, g), mr)];
      b = *(const v16bf*)(drp + ((((size_t)(c * 2 + s) * 6 + wave) * 32 + lane) << 4));
      acc = __builtin_amdgcn_wmma_f32_16x16x32_bf16(false, a, false, b, (short)0, acc, false, false);
    }
    __syncthreads();
  }
#else
  for (int c = 0; c < DIM; ++c){
    __syncthreads();
    for (int idx = tid; idx < 1024; idx += 192){
      int xq = idx % nx, rem = idx / nx, yq = rem % ny, zq = rem / ny;
      sA[0][idx] = x[(size_t)c * 262144 + base + xq + yq * 64 + zq * 4096];
    }
    __syncthreads();
#pragma unroll
    for (int s = 0; s < 2; ++s){
      v16bf a, b;
#pragma unroll
      for (int e = 0; e < 16; ++e)
        a[e] = (bf16_t)sA[0][lidx<PERM>(s * 32 + kfrag(e, g), mr)];
      b = *(const v16bf*)(drp + ((((size_t)(c * 2 + s) * 6 + wave) * 32 + lane) << 4));
      acc = __builtin_amdgcn_wmma_f32_16x16x32_bf16(false, a, false, b, (short)0, acc, false, false);
    }
  }
#endif

#pragma unroll
  for (int j = 0; j < 8; ++j){
    int tok = tile * 16 + j + g * 8;
    xf[tok * DIM + wave * 16 + mr] = acc[j];
  }
}

// ---------------- block reduction over 96 lanes ----------------
__device__ __forceinline__ float reduce96(float v, float* red, int tid){
  red[tid] = v;
  if (tid < 32) red[96 + tid] = 0.f;
  __syncthreads();
  for (int off = 64; off > 0; off >>= 1){
    if (tid < off) red[tid] += red[tid + off];
    __syncthreads();
  }
  float r = red[0];
  __syncthreads();
  return r;
}

// ---------------- LN1 + in-projection (96 -> 384), split xc_pre / silu(z) ----------------
__global__ void k_tok_a(const float* __restrict__ xf, const float* __restrict__ g1,
                        const float* __restrict__ b1, const float* __restrict__ inW,
                        float* __restrict__ xc_pre, float* __restrict__ zs){
  __shared__ float red[128];
  __shared__ float sln[DIM];
  const int t = blockIdx.x, n = threadIdx.x;   // 96 threads
  float v  = xf[t * DIM + n];
  float mu = reduce96(v, red, n) * (1.f / DIM);
  float s2 = reduce96(v * v, red, n) * (1.f / DIM);
  float rs = rsqrtf(s2 - mu * mu + EPS);
  sln[n] = (v - mu) * rs * g1[n] + b1[n];
  __syncthreads();
#pragma unroll
  for (int m = 0; m < 4; ++m){
    int j = m * DIM + n;
    float acc = 0.f;
    for (int c = 0; c < DIM; ++c) acc += sln[c] * inW[c * (2 * DIc) + j];
    if (j < DIc) xc_pre[t * DIc + j] = acc;
    else         zs[t * DIc + (j - DIc)] = silu_f(acc);
  }
}

// ---------------- causal depthwise conv + silu, x-proj (192->38), dt (6->192) ----------------
__global__ void k_tok_b(const float* __restrict__ xc_pre, const float* __restrict__ convW,
                        const float* __restrict__ convB, const float* __restrict__ xpW,
                        const float* __restrict__ dtW, const float* __restrict__ dtB,
                        float* __restrict__ xc, float* __restrict__ dtb,
                        float* __restrict__ Bmb, float* __restrict__ Cmb){
  __shared__ float sxc[DIc];
  __shared__ float sproj[DTR + 2 * DSc];
  const int t = blockIdx.x, d = threadIdx.x;   // 192 threads
  float acc = convB[d];
#pragma unroll
  for (int q = 0; q < DCc; ++q){
    int ts = t - (DCc - 1) + q;
    if (ts >= 0) acc += convW[d * DCc + q] * xc_pre[ts * DIc + d];
  }
  float xv = silu_f(acc);
  sxc[d] = xv;
  xc[t * DIc + d] = xv;
  __syncthreads();
  if (d < DTR + 2 * DSc){
    float p = 0.f;
    for (int c = 0; c < DIc; ++c) p += sxc[c] * xpW[c * (DTR + 2 * DSc) + d];
    sproj[d] = p;
  }
  __syncthreads();
  float s = dtB[d];
#pragma unroll
  for (int r = 0; r < DTR; ++r) s += sproj[r] * dtW[r * DIc + d];
  dtb[t * DIc + d] = softplus_f(s);
  if (d < DSc){
    Bmb[t * DSc + d] = sproj[DTR + d];
    Cmb[t * DSc + d] = sproj[DTR + DSc + d];
  }
}

// ---------------- sequential selective-scan: one block, state (192x16) in registers ----------------
__global__ void k_scan(const float* __restrict__ Alog, const float* __restrict__ Dsk,
                       const float* __restrict__ dtb, const float* __restrict__ xc,
                       const float* __restrict__ Bmb, const float* __restrict__ Cmb,
                       float* __restrict__ ybuf){
  __shared__ float sB[DSc], sC[DSc];
  const int d = threadIdx.x;   // 192 threads
  float hreg[DSc], An[DSc];
#pragma unroll
  for (int s = 0; s < DSc; ++s){ hreg[s] = 0.f; An[s] = -__expf(Alog[d * DSc + s]); }
  const float Dv = Dsk[d];
  for (int t = 0; t < L; ++t){
    if (d < DSc)          sB[d]       = Bmb[t * DSc + d];
    else if (d < 2 * DSc) sC[d - DSc] = Cmb[t * DSc + (d - DSc)];
    __syncthreads();
    float dtv = dtb[t * DIc + d];
    float xv  = xc[t * DIc + d];
    float y = 0.f;
#pragma unroll
    for (int s = 0; s < DSc; ++s){
      hreg[s] = __expf(dtv * An[s]) * hreg[s] + dtv * sB[s] * xv;
      y += hreg[s] * sC[s];
    }
    ybuf[t * DIc + d] = y + Dv * xv;
    __syncthreads();
  }
}

// ---------------- gate * out-proj, LN2 + residual, MLP, emit bf16 ----------------
__global__ void k_post(const float* __restrict__ ybuf, const float* __restrict__ zs,
                       const float* __restrict__ outW, const float* __restrict__ g2,
                       const float* __restrict__ b2, const float* __restrict__ xf,
                       const float* __restrict__ W1, const float* __restrict__ bb1,
                       const float* __restrict__ W2, const float* __restrict__ bb2,
                       bf16_t* __restrict__ outbf){
  __shared__ float yz[DIc];
  __shared__ float mos[DIM];
  __shared__ float hdd[4 * DIM];
  __shared__ float red[128];
  const int t = blockIdx.x, n = threadIdx.x;   // 96 threads
  yz[n]      = ybuf[t * DIc + n]      * zs[t * DIc + n];
  yz[n + 96] = ybuf[t * DIc + n + 96] * zs[t * DIc + n + 96];
  __syncthreads();
  float xm = 0.f;
  for (int dd = 0; dd < DIc; ++dd) xm += yz[dd] * outW[dd * DIM + n];
  float mu = reduce96(xm, red, n) * (1.f / DIM);
  float s2 = reduce96(xm * xm, red, n) * (1.f / DIM);
  float rs = rsqrtf(s2 - mu * mu + EPS);
  float mo = (xm - mu) * rs * g2[n] + b2[n] + xf[t * DIM + n];
  mos[n] = mo;
  __syncthreads();
#pragma unroll
  for (int m = 0; m < 4; ++m){
    int j = m * DIM + n;
    float hv = bb1[j];
    for (int c = 0; c < DIM; ++c) hv += mos[c] * W1[c * (4 * DIM) + j];
    hdd[j] = gelu_f(hv);
  }
  __syncthreads();
  float ov = bb2[n] + mo;
  for (int j = 0; j < 4 * DIM; ++j) ov += hdd[j] * W2[j * DIM + n];
  outbf[t * DIM + n] = (bf16_t)ov;
}

// ---------------- de GEMM: (out @ de_W) + x_perm, scatter-accumulate into d_out ----------------
template<int PERM>
__global__ void k_de_gemm(const bf16_t* __restrict__ outbf, const bf16_t* __restrict__ dep,
                          const float* __restrict__ x, float* __restrict__ out){
  const int lane = threadIdx.x;        // one wave32 per block
  const int g    = lane >> 4;
  const int mr   = lane & 15;
  const int tile = blockIdx.x;         // 256 token tiles
  const int nbt  = blockIdx.y;         // 384 N tiles
  const int t    = tile * 16 + mr;
  // prefetch the strided residual-gather lines while the GEMM runs
#pragma unroll
  for (int j = 0; j < 8; ++j)
    __builtin_prefetch(&x[gidx<PERM>(tile * 16 + j + g * 8, nbt * 16 + mr)], 0, 0);
  const bf16_t* arow = outbf + (size_t)t * DIM;
  v8f acc = {};
#pragma unroll
  for (int kb = 0; kb < 3; ++kb){
    v8bf lo = *(const v8bf*)(arow + kb * 32 + g * 8);
    v8bf hi = *(const v8bf*)(arow + kb * 32 + 16 + g * 8);
    v16bf a = __builtin_shufflevector(lo, hi, 0, 1, 2, 3, 4, 5, 6, 7,
                                      8, 9, 10, 11, 12, 13, 14, 15);
    v16bf b = *(const v16bf*)(dep + ((((size_t)kb * 384 + nbt) * 32 + lane) << 4));
    acc = __builtin_amdgcn_wmma_f32_16x16x32_bf16(false, a, false, b, (short)0, acc, false, false);
  }
#pragma unroll
  for (int j = 0; j < 8; ++j){
    int tok = tile * 16 + j + g * 8;
    int f   = nbt * 16 + mr;
    float val = acc[j] + x[gidx<PERM>(tok, f)];
    out[gidx<0>(tok, f)] += val;   // unique (tok,f) per kernel; perms run sequentially
  }
}

// ---------------- final per-channel layernorm over 64^3 ----------------
__global__ void k_stats(const float* __restrict__ out, float* __restrict__ st){
  __shared__ float r1[256], r2[256];
  const int c = blockIdx.x, tid = threadIdx.x;
  const float* p = out + (size_t)c * VOL;
  float s = 0.f, s2 = 0.f;
  for (int idx = tid; idx < VOL; idx += 256){ float v = p[idx]; s += v; s2 += v * v; }
  r1[tid] = s; r2[tid] = s2; __syncthreads();
  for (int off = 128; off > 0; off >>= 1){
    if (tid < off){ r1[tid] += r1[tid + off]; r2[tid] += r2[tid + off]; }
    __syncthreads();
  }
  if (tid == 0){
    float mu  = r1[0] / VOL;
    float var = r2[0] / VOL - mu * mu;
    st[c] = mu; st[DIM + c] = rsqrtf(var + EPS);
  }
}

__global__ void k_norm(float* __restrict__ out, const float* __restrict__ st){
  int idx = blockIdx.x * 256 + threadIdx.x;
  const int total = DIM * VOL;
  for (; idx < total; idx += gridDim.x * 256){
    int c = idx / VOL;
    out[idx] = (out[idx] - st[c]) * st[DIM + c];
  }
}

} // namespace

extern "C" void kernel_launch(void* const* d_in, const int* in_sizes, int n_in,
                              void* d_out, int out_size, void* d_ws, size_t ws_size,
                              hipStream_t stream) {
  (void)in_sizes; (void)n_in; (void)out_size; (void)ws_size;
  const int li = NL - 1;   // only the last layer contributes (xo is overwritten per iteration)

  const float* x     = (const float*)d_in[0];
  const float* drW   = (const float*)d_in[1]  + (size_t)li * INDIM * DIM;
  const float* deW   = (const float*)d_in[2]  + (size_t)li * DIM * INDIM;
  const float* n1g   = (const float*)d_in[3]  + (size_t)li * DIM;
  const float* n1b   = (const float*)d_in[4]  + (size_t)li * DIM;
  const float* n2g   = (const float*)d_in[5]  + (size_t)li * DIM;
  const float* n2b   = (const float*)d_in[6]  + (size_t)li * DIM;
  const float* inW   = (const float*)d_in[7]  + (size_t)li * DIM * 2 * DIc;
  const float* convW = (const float*)d_in[8]  + (size_t)li * DIc * DCc;
  const float* convB = (const float*)d_in[9]  + (size_t)li * DIc;
  const float* xpW   = (const float*)d_in[10] + (size_t)li * DIc * (DTR + 2 * DSc);
  const float* dtW   = (const float*)d_in[11] + (size_t)li * DTR * DIc;
  const float* dtB   = (const float*)d_in[12] + (size_t)li * DIc;
  const float* Alog  = (const float*)d_in[13] + (size_t)li * DIc * DSc;
  const float* Dsk   = (const float*)d_in[14] + (size_t)li * DIc;
  const float* outW  = (const float*)d_in[15] + (size_t)li * DIc * DIM;
  const float* W1    = (const float*)d_in[16] + (size_t)li * DIM * 4 * DIM;
  const float* b1m   = (const float*)d_in[17] + (size_t)li * 4 * DIM;
  const float* W2    = (const float*)d_in[18] + (size_t)li * 4 * DIM * DIM;
  const float* b2m   = (const float*)d_in[19] + (size_t)li * DIM;
  float* outp = (float*)d_out;

  // ---- workspace carve (~21 MB) ----
  char* base = (char*)d_ws; size_t off = 0;
  auto carve = [&](size_t bytes) -> void* {
    void* p = base + off; off += (bytes + 255) & ~(size_t)255; return p;
  };
  bf16_t* drp    = (bf16_t*)carve((size_t)INDIM * DIM * sizeof(bf16_t));
  bf16_t* dep    = (bf16_t*)carve((size_t)INDIM * DIM * sizeof(bf16_t));
  bf16_t* outbf  = (bf16_t*)carve((size_t)L * DIM * sizeof(bf16_t));
  float*  xf     = (float*)carve((size_t)L * DIM * sizeof(float));
  float*  xc_pre = (float*)carve((size_t)L * DIc * sizeof(float));
  float*  zsb    = (float*)carve((size_t)L * DIc * sizeof(float));
  float*  xcb    = (float*)carve((size_t)L * DIc * sizeof(float));
  float*  dtbuf  = (float*)carve((size_t)L * DIc * sizeof(float));
  float*  ybuf   = (float*)carve((size_t)L * DIc * sizeof(float));
  float*  Bmb    = (float*)carve((size_t)L * DSc * sizeof(float));
  float*  Cmb    = (float*)carve((size_t)L * DSc * sizeof(float));
  float*  st     = (float*)carve((size_t)2 * DIM * sizeof(float));

  const int NOUT = DIM * VOL;
  k_pack<<<(INDIM * DIM + 255) / 256, 256, 0, stream>>>(drW, deW, drp, dep);
  k_zero<<<(NOUT + 255) / 256, 256, 0, stream>>>(outp, NOUT);

#define RUN_PERM(P)                                                                         \
  k_dr_gemm<P><<<256, 192, 0, stream>>>(x, drp, xf);                                        \
  k_tok_a<<<L, DIM, 0, stream>>>(xf, n1g, n1b, inW, xc_pre, zsb);                           \
  k_tok_b<<<L, DIc, 0, stream>>>(xc_pre, convW, convB, xpW, dtW, dtB, xcb, dtbuf, Bmb, Cmb);\
  k_scan<<<1, DIc, 0, stream>>>(Alog, Dsk, dtbuf, xcb, Bmb, Cmb, ybuf);                     \
  k_post<<<L, DIM, 0, stream>>>(ybuf, zsb, outW, n2g, n2b, xf, W1, b1m, W2, b2m, outbf);    \
  k_de_gemm<P><<<dim3(256, 384), 32, 0, stream>>>(outbf, dep, x, outp);

  RUN_PERM(0)
  RUN_PERM(1)
  RUN_PERM(2)
#undef RUN_PERM

  k_stats<<<DIM, 256, 0, stream>>>(outp, st);
  k_norm<<<4096, 256, 0, stream>>>(outp, st);
}
